// MaskMultiHeadedSelfAttention_63213328663256
// MI455X (gfx1250) — matrix operand order repaired and checked by reading, not compiled, not run on hardware
//
#include <hip/hip_runtime.h>
#include <hip/hip_bf16.h>
#include <cstdint>

// ---------------- CDNA5 WMMA types ----------------
typedef __attribute__((ext_vector_type(16))) __bf16 v16bf;
typedef __attribute__((ext_vector_type(8)))  float  v8f;
typedef __attribute__((ext_vector_type(4)))  unsigned int v4u;

union ABfrag { v16bf bf; v4u u[2]; };   // 32 bytes: one 16-bit A/B operand (8 VGPRs)

#define NHEAD  12
#define BATCH  2
#define NMASK  4
#define SEQ    1025
#define SPAD   1056              // 66 tiles of 16 -> 33 pairs of 32 keys
#define DMODEL 768
#define HDIM   64
#define ROWS   (BATCH*SEQ)       // 2050
#define ROWSP  2064              // 129 row-tiles for the QKV GEMM
#define KTILES 24                // 768 / 32
#define NKT    66                // SPAD / 16

// ---------------------------------------------------------------------------
// prep: fp32 -> bf16 conversion of x (zero-padded rows) and Wq|Wk|Wv,
//       plus zero-fill of the padded Q / K / V^T buffers.
// ---------------------------------------------------------------------------
__global__ __launch_bounds__(256) void prep_kernel(
    const float* __restrict__ x,
    const float* __restrict__ Wq, const float* __restrict__ Wk,
    const float* __restrict__ Wv,
    __bf16* __restrict__ xbf, __bf16* __restrict__ wbf,
    __bf16* __restrict__ zbuf /* Qbf|Kbf|Vt contiguous */)
{
  const unsigned XBF_E = ROWSP*DMODEL;
  const unsigned WBF_E = 3u*DMODEL*DMODEL;
  const unsigned QKV_E = BATCH*NHEAD*SPAD*HDIM;
  unsigned i = blockIdx.x*256u + threadIdx.x;
  if (i < XBF_E) {
    unsigned row = i / DMODEL;
    float v = (row < ROWS) ? x[i] : 0.0f;
    xbf[i] = (__bf16)v;
    return;
  }
  i -= XBF_E;
  if (i < WBF_E) {
    unsigned z = i / (DMODEL*DMODEL);
    unsigned r = i - z*(DMODEL*DMODEL);
    const float* W = (z==0) ? Wq : ((z==1) ? Wk : Wv);
    wbf[i] = (__bf16)W[r];
    return;
  }
  i -= WBF_E;
  if (i < 3u*QKV_E) zbuf[i] = (__bf16)0.0f;
}

// ---------------------------------------------------------------------------
// qkv: out = x @ W^T + b, one wave per 16x16 tile, bf16 WMMA, fp32 acc.
//      z = 0/1/2 -> Q / K / V.  Q,K stored (b,h,SPAD,hd); V stored transposed
//      (b,h,hd,SPAD) so P@V B-fragments are contiguous later.
// ---------------------------------------------------------------------------
__global__ __launch_bounds__(32) void qkv_kernel(
    const __bf16* __restrict__ xbf, const __bf16* __restrict__ wbf,
    const float* __restrict__ bq, const float* __restrict__ bk,
    const float* __restrict__ bv,
    __bf16* __restrict__ Qbf, __bf16* __restrict__ Kbf, __bf16* __restrict__ Vt)
{
  const int lane = threadIdx.x;
  const int n_l = lane & 15, half = lane >> 4;
  const int ntile = blockIdx.x;          // 0..47
  const int mtile = blockIdx.y;          // 0..128
  const int z     = blockIdx.z;          // 0..2
  const __bf16* W    = wbf + (size_t)z * DMODEL * DMODEL;
  const __bf16* arow = xbf + (size_t)(mtile*16 + n_l) * DMODEL;
  const __bf16* bcol = W   + (size_t)(ntile*16 + n_l) * DMODEL + half*16;

  v8f acc = {};
  for (int kt = 0; kt < KTILES; ++kt) {
    const int kb = kt*32;
    ABfrag a, bfr;
    // A (16x32 bf16): lane m=n_l, K = half*16 + 0..15
    a.u[0] = *(const v4u*)(arow + kb + half*16);
    a.u[1] = *(const v4u*)(arow + kb + half*16 + 8);
    // B (32x16 bf16): lane n=n_l, K = half*16 + 0..15 (row-major W == col-major B)
    bfr.u[0] = *(const v4u*)(bcol + kb);
    bfr.u[1] = *(const v4u*)(bcol + kb + 8);
    acc = __builtin_amdgcn_wmma_f32_16x16x32_bf16(false, a.bf, false, bfr.bf,
                                                  (short)0, acc, false, false);
  }

  const int col  = ntile*16 + n_l;
  const float* bias = (z==0) ? bq : ((z==1) ? bk : bv);
  const float bval = bias[col];
  const int head = col >> 6, d = col & 63;
#pragma unroll
  for (int r = 0; r < 8; ++r) {
    const int row = mtile*16 + r + half*8;   // C layout: M = r + 8*(lane>>4)
    if (row >= ROWS) continue;
    const int bb = row / SEQ;
    const int s  = row - bb*SEQ;
    const __bf16 v = (__bf16)(acc[r] + bval);
    const size_t bh = (size_t)(bb*NHEAD + head);
    if (z == 0)      Qbf[(bh*SPAD + s)*HDIM + d] = v;
    else if (z == 1) Kbf[(bh*SPAD + s)*HDIM + d] = v;
    else             Vt [(bh*HDIM + d)*SPAD + s] = v;
  }
}

// ---------------------------------------------------------------------------
// attn: flash-style attention; one wave per (qtile, b, variant, head).
//       variant 0 = unmasked -> h; variants 1..4 -> new_h with mask bias.
// ---------------------------------------------------------------------------
__global__ __launch_bounds__(32) void attn_kernel(
    const __bf16* __restrict__ Qbf, const __bf16* __restrict__ Kbf,
    const __bf16* __restrict__ Vt,  const int* __restrict__ masks,
    float* __restrict__ out_h, float* __restrict__ out_nh)
{
  __shared__ __align__(16) __bf16 plds[16][32];   // prob tile, C->A layout hop

  const int lane = threadIdx.x;
  const int n_l = lane & 15, half = lane >> 4;
  const int qbase = blockIdx.x * 16;
  int comb = blockIdx.y;
  const int head = comb % NHEAD;  comb /= NHEAD;
  const int variant = comb % 5;   const int b = comb / 5;

  const __bf16* Qh = Qbf + (size_t)(b*NHEAD + head) * SPAD * HDIM;
  const __bf16* Kh = Kbf + (size_t)(b*NHEAD + head) * SPAD * HDIM;
  const __bf16* Vh = Vt  + (size_t)(b*NHEAD + head) * HDIM * SPAD;

  // Q A-fragments for the whole hd=64 (two K=32 chunks), resident in regs.
  ABfrag aq0, aq1;
  {
    const __bf16* p = Qh + (size_t)(qbase + n_l) * HDIM;
    aq0.u[0] = *(const v4u*)(p      + half*16);
    aq0.u[1] = *(const v4u*)(p + 8  + half*16);
    aq1.u[0] = *(const v4u*)(p + 32 + half*16);
    aq1.u[1] = *(const v4u*)(p + 40 + half*16);
  }

  v8f O0 = {}, O1 = {}, O2 = {}, O3 = {};
  float mrun[8], lrun[8];
#pragma unroll
  for (int r = 0; r < 8; ++r) { mrun[r] = -1e30f; lrun[r] = 0.0f; }

  for (int jp = 0; jp < NKT/2; ++jp) {
#pragma unroll
    for (int sub = 0; sub < 2; ++sub) {
      const int jbase = (jp*2 + sub) * 16;
      // K B-fragments: lane n=n_l picks key row jbase+n_l; contiguous in d.
      ABfrag bk0, bk1;
      const __bf16* kp = Kh + (size_t)(jbase + n_l) * HDIM + half*16;
      bk0.u[0] = *(const v4u*)(kp);       bk0.u[1] = *(const v4u*)(kp + 8);
      bk1.u[0] = *(const v4u*)(kp + 32);  bk1.u[1] = *(const v4u*)(kp + 40);

      v8f s = {};
      s = __builtin_amdgcn_wmma_f32_16x16x32_bf16(false, aq0.bf, false, bk0.bf,
                                                  (short)0, s, false, false);
      s = __builtin_amdgcn_wmma_f32_16x16x32_bf16(false, aq1.bf, false, bk1.bf,
                                                  (short)0, s, false, false);

      // additive mask bias (per key column; same for all 8 rows in this lane)
      const int col = jbase + n_l;
      float bias;
      if (col >= SEQ)            bias = -1e30f;     // padding: fully excluded
      else if (variant == 0)     bias = 0.0f;
      else if (col == 0)         bias = -10000.0f;  // CLS padded with mask=0
      else bias = masks[((b*NMASK + (variant-1)) << 10) + (col-1)] ? 0.0f
                                                                   : -10000.0f;
      // online softmax; row M = r + 8*half lives in one 16-lane half.
#pragma unroll
      for (int r = 0; r < 8; ++r) {
        float v = s[r]*0.125f + bias;                  // scale = 1/sqrt(64)
        float mx = v;
        mx = fmaxf(mx, __shfl_xor(mx, 1, 16));
        mx = fmaxf(mx, __shfl_xor(mx, 2, 16));
        mx = fmaxf(mx, __shfl_xor(mx, 4, 16));
        mx = fmaxf(mx, __shfl_xor(mx, 8, 16));
        const float mnew = fmaxf(mrun[r], mx);
        const float corr = __expf(mrun[r] - mnew);
        const float p    = __expf(v - mnew);
        float ps = p;
        ps += __shfl_xor(ps, 1, 16);
        ps += __shfl_xor(ps, 2, 16);
        ps += __shfl_xor(ps, 4, 16);
        ps += __shfl_xor(ps, 8, 16);
        lrun[r] = lrun[r]*corr + ps;
        mrun[r] = mnew;
        O0[r] *= corr; O1[r] *= corr; O2[r] *= corr; O3[r] *= corr;
        plds[r + half*8][sub*16 + n_l] = (__bf16)p;   // C-layout -> LDS tile
      }
    }
    __syncthreads();  // single-wave WG: lowers to S_NOP; DS is in-order anyway

    // P (16x32 bf16) A-fragment back from LDS
    ABfrag pa;
    const __bf16* pr = &plds[n_l][0];
    pa.u[0] = *(const v4u*)(pr + half*16);
    pa.u[1] = *(const v4u*)(pr + half*16 + 8);

    // O += P @ V : V^T row d is contiguous over keys -> 16B B-fragment loads
    const __bf16* vp = Vh + jp*32 + half*16;
    {
      ABfrag bv0;
      const __bf16* q = vp + (size_t)(0*16 + n_l) * SPAD;
      bv0.u[0] = *(const v4u*)q; bv0.u[1] = *(const v4u*)(q + 8);
      O0 = __builtin_amdgcn_wmma_f32_16x16x32_bf16(false, pa.bf, false, bv0.bf,
                                                   (short)0, O0, false, false);
    }
    {
      ABfrag bv1;
      const __bf16* q = vp + (size_t)(1*16 + n_l) * SPAD;
      bv1.u[0] = *(const v4u*)q; bv1.u[1] = *(const v4u*)(q + 8);
      O1 = __builtin_amdgcn_wmma_f32_16x16x32_bf16(false, pa.bf, false, bv1.bf,
                                                   (short)0, O1, false, false);
    }
    {
      ABfrag bv2;
      const __bf16* q = vp + (size_t)(2*16 + n_l) * SPAD;
      bv2.u[0] = *(const v4u*)q; bv2.u[1] = *(const v4u*)(q + 8);
      O2 = __builtin_amdgcn_wmma_f32_16x16x32_bf16(false, pa.bf, false, bv2.bf,
                                                   (short)0, O2, false, false);
    }
    {
      ABfrag bv3;
      const __bf16* q = vp + (size_t)(3*16 + n_l) * SPAD;
      bv3.u[0] = *(const v4u*)q; bv3.u[1] = *(const v4u*)(q + 8);
      O3 = __builtin_amdgcn_wmma_f32_16x16x32_bf16(false, pa.bf, false, bv3.bf,
                                                   (short)0, O3, false, false);
    }
    __syncthreads();
  }

  // normalize + scatter (h layout: (B,S,D) with head h at cols h*64..h*64+63)
#pragma unroll
  for (int r = 0; r < 8; ++r) {
    const int srow = qbase + r + half*8;
    if (srow >= SEQ) continue;
    const float inv = 1.0f / lrun[r];
    float* outp;
    const int colbase = head*HDIM + n_l;
    if (variant == 0) {
      outp = out_h + ((size_t)(b*SEQ + srow))*DMODEL + colbase;
    } else {
      const int bm = b*NMASK + variant - 1;
      outp = out_nh + ((size_t)(bm*SEQ + srow))*DMODEL + colbase;
    }
    outp[0]  = O0[r]*inv;
    outp[16] = O1[r]*inv;
    outp[32] = O2[r]*inv;
    outp[48] = O3[r]*inv;
  }
}

// ---------------------------------------------------------------------------
// tail: img_idx = repeat(arange(B), M) and new_x = broadcast copy of x.
// ---------------------------------------------------------------------------
__global__ __launch_bounds__(256) void tail_kernel(
    const float* __restrict__ x, float* __restrict__ out_idx,
    float* __restrict__ out_nx)
{
  const unsigned per = SEQ*DMODEL;
  const unsigned total = BATCH*NMASK*per;
  unsigned i = blockIdx.x*256u + threadIdx.x;
  if (i < BATCH*NMASK) out_idx[i] = (float)(i / NMASK);
  if (i < total) {
    const unsigned bm  = i / per;
    const unsigned rem = i - bm*per;
    out_nx[i] = x[(bm/NMASK)*per + rem];
  }
}

// ---------------------------------------------------------------------------
extern "C" void kernel_launch(void* const* d_in, const int* in_sizes, int n_in,
                              void* d_out, int out_size, void* d_ws, size_t ws_size,
                              hipStream_t stream) {
  (void)in_sizes; (void)n_in; (void)out_size; (void)ws_size;
  const float* x   = (const float*)d_in[0];
  const int*   msk = (const int*)  d_in[1];
  const float* Wq  = (const float*)d_in[2];
  const float* bq  = (const float*)d_in[3];
  const float* Wk  = (const float*)d_in[4];
  const float* bk  = (const float*)d_in[5];
  const float* Wv  = (const float*)d_in[6];
  const float* bv  = (const float*)d_in[7];
  float* out = (float*)d_out;

  const size_t XBF_E = (size_t)ROWSP*DMODEL;          // padded x, bf16
  const size_t WBF_E = 3ull*DMODEL*DMODEL;            // Wq|Wk|Wv, bf16
  const size_t QKV_E = (size_t)BATCH*NHEAD*SPAD*HDIM; // per Q/K/Vt buffer
  uint8_t* w = (uint8_t*)d_ws;
  __bf16* xbf = (__bf16*)w;
  __bf16* wbf = (__bf16*)(w + XBF_E*2);
  __bf16* Qbf = (__bf16*)(w + (XBF_E + WBF_E)*2);
  __bf16* Kbf = (__bf16*)(w + (XBF_E + WBF_E + QKV_E)*2);
  __bf16* Vt  = (__bf16*)(w + (XBF_E + WBF_E + 2*QKV_E)*2);

  float* out_h   = out;
  float* out_nh  = out + (size_t)BATCH*SEQ*DMODEL;
  float* out_idx = out_nh + (size_t)BATCH*NMASK*SEQ*DMODEL;
  float* out_nx  = out_idx + BATCH*NMASK;

  {
    const unsigned total = (unsigned)(XBF_E + WBF_E + 3*QKV_E);
    prep_kernel<<<dim3((total + 255)/256), dim3(256), 0, stream>>>(
        x, Wq, Wk, Wv, xbf, wbf, Qbf);
  }
  qkv_kernel<<<dim3(DMODEL/16, ROWSP/16, 3), dim3(32), 0, stream>>>(
      xbf, wbf, bq, bk, bv, Qbf, Kbf, Vt);
  attn_kernel<<<dim3(65, BATCH*5*NHEAD), dim3(32), 0, stream>>>(
      Qbf, Kbf, Vt, msk, out_h, out_nh);
  {
    const unsigned total = (unsigned)(BATCH*NMASK)*SEQ*DMODEL;
    tail_kernel<<<dim3((total + 255)/256), dim3(256), 0, stream>>>(
        x, out_idx, out_nx);
  }
}
